// AttnDecoderRNN_42743514530240
// MI455X (gfx1250) — matrix operand order repaired and checked by reading, not compiled, not run on hardware
//
#include <hip/hip_runtime.h>
#include <math.h>

// ---------------------------------------------------------------------------
// AttnDecoderRNN for MI455X (gfx1250, wave32, WMMA).
//
// Strategy (compile-only reasoning, see analysis):
//  * Everything is batch-1 GEMV -> bandwidth bound.  Convert all weights to
//    f16 once per launch (zero-padded K to mult-of-32, N to mult-of-16), then
//    run every matmul through one generic v_wmma_f32_16x16x32_f16 GEMV kernel
//    (16 output rows per wave in the A/M dimension, activation vector in
//    B column 0, f32 accumulation).  f16 weights (~57MB) fit in the 192MB L2.
//  * Serial glue (embedding/concat, masked softmax + context, GRU gates) runs
//    in tiny fp32 kernels; 19 decode steps are expressed as a launch sequence
//    (graph replay amortizes launch cost).
// ---------------------------------------------------------------------------

typedef _Float16 half_t;
typedef __attribute__((ext_vector_type(16))) _Float16 v16h;
typedef __attribute__((ext_vector_type(8)))  _Float16 v8h;
typedef __attribute__((ext_vector_type(8)))  float    v8f;

union V16H { v16h v; v8h h2[2]; };

#define V_SZ   50257
#define EMB_SZ 300
#define H_SZ   512
#define HS_SZ  256
#define BH_SZ  256
#define NP_SZ  256
#define ML_SZ  50
#define STEPS  19
#define CAT_KP 832   // 812 padded to mult of 32

// ---------------------------------------------------------------------------
// fp32 -> f16 weight conversion with K-padding (to Kp) and N-padding (to Np).
// ---------------------------------------------------------------------------
__global__ void cvt_pad_kernel(const float* __restrict__ src, half_t* __restrict__ dst,
                               int N, int K, int Np, int Kp) {
    size_t total = (size_t)Np * (size_t)Kp;
    for (size_t i = (size_t)blockIdx.x * blockDim.x + threadIdx.x; i < total;
         i += (size_t)gridDim.x * blockDim.x) {
        int r = (int)(i / (size_t)Kp);
        int k = (int)(i % (size_t)Kp);
        float v = (r < N && k < K) ? src[(size_t)r * K + k] : 0.0f;
        dst[i] = (half_t)v;
    }
}

// ---------------------------------------------------------------------------
// Generic WMMA GEMV:  out[r] = act( sum_k W[r][k] * x[k] + bias[r] )
// One wave computes a 16-row tile.  A = 16x32 f16 weight tile (per-lane layout
// from CDNA5 ISA 7.12.2: lane L holds row L%16, K-halves chosen by L/16).
// B = activation chunk in column 0 only (lane 0: K=0..7,16..23, lane 16: K=8..15,24..31).
// D(f32) column 0 lands in lanes 0/16, VGPRs 0..7.
// ---------------------------------------------------------------------------
__global__ void wmma_gemv_kernel(const half_t* __restrict__ W, const half_t* __restrict__ x,
                                 int Kp, int Nvalid, const float* __restrict__ bias,
                                 float* __restrict__ outF, half_t* __restrict__ outH,
                                 int doRelu) {
    const int lane = threadIdx.x & 31;
    const int wave = blockIdx.x * (blockDim.x >> 5) + (threadIdx.x >> 5);
    const int rowBase = wave << 4;
    if (rowBase >= Nvalid) return;                 // wave-uniform exit

    const int m  = lane & 15;
    const int kh = (lane >> 4) << 3;               // 0 or 8: K-half select
    const half_t* wrow = W + (size_t)(rowBase + m) * (size_t)Kp + kh;

    const v16h zerov = {};
    v8f acc = {};
    for (int k = 0; k < Kp; k += 32) {
        V16H a;
        a.h2[0] = *(const v8h*)(wrow + k);         // half slots 0-7 : K = k+kh+0..7
        a.h2[1] = *(const v8h*)(wrow + k + 16);    // half slots 8-15: K = k+16+kh+0..7
        V16H xu;
        xu.h2[0] = *(const v8h*)(x + k + kh);
        xu.h2[1] = *(const v8h*)(x + k + 16 + kh);
        v16h xv = xu.v;
        v16h b  = ((lane & 15) == 0) ? xv : zerov; // only column N==0 carries data
        acc = __builtin_amdgcn_wmma_f32_16x16x32_f16(
            /*neg_a=*/false, a.v, /*neg_b=*/false, b,
            /*c_mod=*/(short)0, acc, /*reuse_a=*/false, /*reuse_b=*/false);
    }

    if ((lane & 15) == 0) {                        // lanes 0 and 16 hold N==0
        int rOff = (lane >> 4) << 3;               // lane0 -> rows +0..7, lane16 -> +8..15
        for (int r = 0; r < 8; ++r) {
            int row = rowBase + rOff + r;
            if (row < Nvalid) {
                float y = acc[r] + (bias ? bias[row] : 0.0f);
                if (doRelu) y = y > 0.0f ? y : 0.0f;
                if (outF) outF[row] = y;
                if (outH) outH[row] = (half_t)y;
            }
        }
    }
}

// ---------------------------------------------------------------------------
// Glue kernels (fp32, single block).
// ---------------------------------------------------------------------------
__global__ void init_h_kernel(const float* __restrict__ hidden,
                              float* __restrict__ h, half_t* __restrict__ h16) {
    int i = threadIdx.x + blockIdx.x * blockDim.x;
    if (i < H_SZ) { float v = hidden[i]; h[i] = v; h16[i] = (half_t)v; }
}

// x = relu(emb[tok]); cat16 = [x(300), h(512)] padded to 832
__global__ void embed_cat_kernel(const int* __restrict__ input_tok,
                                 const int* __restrict__ answer, int t,
                                 const float* __restrict__ emb,
                                 const float* __restrict__ h,
                                 float* __restrict__ x, half_t* __restrict__ cat16) {
    int tok = (t == 0) ? input_tok[0] : answer[t];
    for (int i = threadIdx.x; i < CAT_KP; i += blockDim.x) {
        float v;
        if (i < EMB_SZ) {
            v = emb[(size_t)tok * EMB_SZ + i];
            v = v > 0.0f ? v : 0.0f;
            x[i] = v;
        } else if (i < EMB_SZ + H_SZ) {
            v = h[i - EMB_SZ];
        } else {
            v = 0.0f;
        }
        cat16[i] = (half_t)v;
    }
}

// masked softmax over n scores (mask==1 -> -1e9, per reference), then
// ctx[j] = sum_i a[i] * feat[i][j]  (featDim = 256); also emit a[] to d_out.
__global__ void softmax_ctx_kernel(const float* __restrict__ scores,
                                   const int* __restrict__ mask, int n,
                                   const float* __restrict__ feat,
                                   float* __restrict__ ctx,
                                   float* __restrict__ attn_out) {
    __shared__ float a[256];
    __shared__ float red[256];
    int tid = threadIdx.x;

    float s = -1e30f;
    if (tid < n) {
        float v = mask[tid] ? -1e9f : scores[tid];
        a[tid] = v;
        s = v;
    }
    red[tid] = s;
    __syncthreads();
    for (int off = 128; off > 0; off >>= 1) {
        if (tid < off) red[tid] = fmaxf(red[tid], red[tid + off]);
        __syncthreads();
    }
    float mx = red[0];
    __syncthreads();

    float e = 0.0f;
    if (tid < n) { e = expf(a[tid] - mx); a[tid] = e; }
    red[tid] = e;
    __syncthreads();
    for (int off = 128; off > 0; off >>= 1) {
        if (tid < off) red[tid] += red[tid + off];
        __syncthreads();
    }
    float inv = 1.0f / red[0];
    __syncthreads();

    if (tid < n) { a[tid] *= inv; attn_out[tid] = a[tid]; }
    __syncthreads();

    float accv = 0.0f;
    for (int i = 0; i < n; ++i) accv += a[i] * feat[(size_t)i * 256 + tid];
    ctx[tid] = accv;
}

// comb16 = [x(300), ctx(256), ctx_op(256)] padded to 832
__global__ void build_comb_kernel(const float* __restrict__ x,
                                  const float* __restrict__ ctx,
                                  const float* __restrict__ ctx_op,
                                  half_t* __restrict__ comb16) {
    for (int i = threadIdx.x; i < CAT_KP; i += blockDim.x) {
        float v;
        if (i < 300)      v = x[i];
        else if (i < 556) v = ctx[i - 300];
        else if (i < 812) v = ctx_op[i - 556];
        else              v = 0.0f;
        comb16[i] = (half_t)v;
    }
}

// GRU gates (torch order r,z,n); updates h (fp32) and h16 in place.
__global__ void gru_gate_kernel(const float* __restrict__ gi, const float* __restrict__ gh,
                                float* __restrict__ h, half_t* __restrict__ h16) {
    int j = threadIdx.x + blockIdx.x * blockDim.x;
    if (j >= H_SZ) return;
    float r  = 1.0f / (1.0f + expf(-(gi[j] + gh[j])));
    float z  = 1.0f / (1.0f + expf(-(gi[H_SZ + j] + gh[H_SZ + j])));
    float nn = tanhf(gi[2 * H_SZ + j] + r * gh[2 * H_SZ + j]);
    float hn = (1.0f - z) * nn + z * h[j];
    h[j] = hn;
    h16[j] = (half_t)hn;
}

// ---------------------------------------------------------------------------
// Host side.
// ---------------------------------------------------------------------------
extern "C" void kernel_launch(void* const* d_in, const int* in_sizes, int n_in,
                              void* d_out, int out_size, void* d_ws, size_t ws_size,
                              hipStream_t stream) {
    (void)in_sizes; (void)n_in; (void)out_size; (void)ws_size;

    // ---- inputs (setup_inputs order) ----
    const int*   input_tok  = (const int*)  d_in[0];
    const float* hidden     = (const float*)d_in[1];
    const float* enc_out    = (const float*)d_in[2];   // [50,256]
    const float* obj_prop   = (const float*)d_in[3];   // [256,256]
    const int*   obj_mask   = (const int*)  d_in[4];
    const int*   lang_mask  = (const int*)  d_in[5];
    const int*   answer     = (const int*)  d_in[6];
    const float* emb        = (const float*)d_in[7];
    const float* attn_w1    = (const float*)d_in[8];
    const float* attn_b1    = (const float*)d_in[9];
    const float* attn_w2    = (const float*)d_in[10];
    const float* attn_b2    = (const float*)d_in[11];
    const float* attn_op_w1 = (const float*)d_in[12];
    const float* attn_op_b1 = (const float*)d_in[13];
    const float* attn_op_w2 = (const float*)d_in[14];
    const float* attn_op_b2 = (const float*)d_in[15];
    const float* cmb_w1     = (const float*)d_in[16];
    const float* cmb_b1     = (const float*)d_in[17];
    const float* cmb_w2     = (const float*)d_in[18];
    const float* cmb_b2     = (const float*)d_in[19];
    const float* gru_w_ih   = (const float*)d_in[20];
    const float* gru_w_hh   = (const float*)d_in[21];
    const float* gru_b_ih   = (const float*)d_in[22];
    const float* gru_b_hh   = (const float*)d_in[23];
    const float* out_w      = (const float*)d_in[24];
    const float* out_b      = (const float*)d_in[25];

    // ---- outputs: logits [19,V], attn_w [19,50], attn_ob [19,256] ----
    float* out_logits = (float*)d_out;
    float* out_attnl  = out_logits + (size_t)STEPS * V_SZ;
    float* out_attno  = out_attnl  + (size_t)STEPS * ML_SZ;

    // ---- workspace carve-out (256B aligned); total ~57.5 MB ----
    char* ws = (char*)d_ws;
    size_t off = 0;
    auto take = [&](size_t bytes) -> void* {
        void* p = ws + off;
        off = (off + bytes + 255) & ~(size_t)255;
        return p;
    };
    float*  h_f32     = (float*) take(H_SZ * 4);
    half_t* h16       = (half_t*)take(H_SZ * 2);
    float*  x_f32     = (float*) take(304 * 4);
    half_t* cat16     = (half_t*)take(CAT_KP * 2);
    half_t* hid_l16   = (half_t*)take(256 * 2);
    half_t* hid_op16  = (half_t*)take(256 * 2);
    float*  scores_l  = (float*) take(64 * 4);
    float*  scores_op = (float*) take(256 * 4);
    float*  ctx       = (float*) take(256 * 4);
    float*  ctx_op    = (float*) take(256 * 4);
    half_t* comb16    = (half_t*)take(CAT_KP * 2);
    half_t* c1_16     = (half_t*)take(512 * 2);
    half_t* g16       = (half_t*)take(512 * 2);
    float*  gi        = (float*) take(1536 * 4);
    float*  gh        = (float*) take(1536 * 4);
    // f16 weights (K padded to mult-32, N padded to mult-16)
    half_t* w_attn1    = (half_t*)take((size_t)256  * CAT_KP * 2);
    half_t* w_attn_op1 = (half_t*)take((size_t)256  * CAT_KP * 2);
    half_t* w_attn2    = (half_t*)take((size_t)64   * 256    * 2);
    half_t* w_attn_op2 = (half_t*)take((size_t)256  * 256    * 2);
    half_t* w_cmb1     = (half_t*)take((size_t)512  * CAT_KP * 2);
    half_t* w_cmb2     = (half_t*)take((size_t)512  * 512    * 2);
    half_t* w_gih      = (half_t*)take((size_t)1536 * 512    * 2);
    half_t* w_ghh      = (half_t*)take((size_t)1536 * 512    * 2);
    half_t* w_out      = (half_t*)take((size_t)50272 * 512   * 2);

    auto cvt = [&](const float* src, half_t* dst, int N, int K, int Np, int Kp) {
        size_t total = (size_t)Np * Kp;
        int blocks = (int)((total + 255) / 256);
        if (blocks > 8192) blocks = 8192;
        cvt_pad_kernel<<<blocks, 256, 0, stream>>>(src, dst, N, K, Np, Kp);
    };
    auto gemv = [&](const half_t* W, const half_t* x, int Kp, int N,
                    const float* bias, float* outF, half_t* outH, int relu) {
        int waves = (N + 15) / 16;
        int blocks, thr;
        if (waves < 8) { blocks = 1; thr = waves * 32; }
        else           { blocks = (waves + 7) / 8; thr = 256; }
        wmma_gemv_kernel<<<blocks, thr, 0, stream>>>(W, x, Kp, N, bias, outF, outH, relu);
    };

    // ---- prologue: weights -> f16 (padded), init hidden state ----
    cvt(attn_w1,    w_attn1,    256,  812, 256,   CAT_KP);
    cvt(attn_op_w1, w_attn_op1, 256,  812, 256,   CAT_KP);
    cvt(attn_w2,    w_attn2,    50,   256, 64,    256);
    cvt(attn_op_w2, w_attn_op2, 256,  256, 256,   256);
    cvt(cmb_w1,     w_cmb1,     512,  812, 512,   CAT_KP);
    cvt(cmb_w2,     w_cmb2,     512,  512, 512,   512);
    cvt(gru_w_ih,   w_gih,      1536, 512, 1536,  512);
    cvt(gru_w_hh,   w_ghh,      1536, 512, 1536,  512);
    cvt(out_w,      w_out,      V_SZ, 512, 50272, 512);
    init_h_kernel<<<1, 512, 0, stream>>>(hidden, h_f32, h16);

    // ---- 19 decode steps ----
    for (int t = 0; t < STEPS; ++t) {
        embed_cat_kernel<<<1, 256, 0, stream>>>(input_tok, answer, t, emb, h_f32,
                                                x_f32, cat16);
        // object attention
        gemv(w_attn_op1, cat16,    CAT_KP, 256, attn_op_b1, nullptr,   hid_op16, 1);
        gemv(w_attn_op2, hid_op16, 256,    256, attn_op_b2, scores_op, nullptr,  0);
        // language attention
        gemv(w_attn1,    cat16,    CAT_KP, 256, attn_b1,    nullptr,   hid_l16,  1);
        gemv(w_attn2,    hid_l16,  256,    50,  attn_b2,    scores_l,  nullptr,  0);
        // softmax + context (also emit attention weights)
        softmax_ctx_kernel<<<1, 256, 0, stream>>>(scores_op, obj_mask, NP_SZ,
                                                  obj_prop, ctx_op,
                                                  out_attno + (size_t)t * NP_SZ);
        softmax_ctx_kernel<<<1, 256, 0, stream>>>(scores_l, lang_mask, ML_SZ,
                                                  enc_out, ctx,
                                                  out_attnl + (size_t)t * ML_SZ);
        // combine MLP
        build_comb_kernel<<<1, 256, 0, stream>>>(x_f32, ctx, ctx_op, comb16);
        gemv(w_cmb1, comb16, CAT_KP, 512, cmb_b1, nullptr, c1_16, 1);
        gemv(w_cmb2, c1_16,  512,    512, cmb_b2, nullptr, g16,   1);
        // GRU cell
        gemv(w_gih, g16, 512, 1536, gru_b_ih, gi, nullptr, 0);
        gemv(w_ghh, h16, 512, 1536, gru_b_hh, gh, nullptr, 0);
        gru_gate_kernel<<<1, 512, 0, stream>>>(gi, gh, h_f32, h16);
        // output projection (the big one: 50257x512, full-chip spread)
        gemv(w_out, h16, 512, V_SZ, out_b, out_logits + (size_t)t * V_SZ, nullptr, 0);
    }
}
